// RelativeMultiHeadAttention_27642409517060
// MI455X (gfx1250) — matrix operand order repaired and checked by reading, not compiled, not run on hardware
//
#include <hip/hip_runtime.h>

// ---------------------------------------------------------------------------
// Relative multi-head attention (Transformer-XL style) for MI455X / gfx1250.
// B=4, S=1024, D=1024, H=16, DK=DH=64.  All GEMMs use v_wmma_f32_16x16x32_bf16
// (fp32 accumulate).  Flash-style attention streams K/V, never spills fp32
// scores; BD is computed via WMMA and scatter-shifted (exact reference
// semantics incl. the wrap artifact) into a bf16 buffer in workspace.
// Workspace use: ~190 MB (xbf/relbf/WT*4/qu/qv/k/v/vT/qrel + 128MB BD-shift).
// ---------------------------------------------------------------------------

#define BB 4
#define SS 1024
#define DD 1024
#define HH 16
#define DKK 64
#define NN (BB*SS)     // 4096
#define HD (HH*DKK)    // 1024

typedef __bf16 bf16;
typedef __attribute__((ext_vector_type(8)))  __bf16 bf16x8;
typedef __attribute__((ext_vector_type(16))) __bf16 v16bf;
typedef __attribute__((ext_vector_type(8)))  float  v8f;

static __device__ __forceinline__ v16bf cat8(bf16x8 lo, bf16x8 hi) {
  return __builtin_shufflevector(lo, hi, 0,1,2,3,4,5,6,7,8,9,10,11,12,13,14,15);
}
static __device__ __forceinline__ v8f zero_v8f() {
  v8f r;
#pragma unroll
  for (int i = 0; i < 8; ++i) r[i] = 0.0f;
  return r;
}
static __device__ __forceinline__ bf16x8 ld8(const bf16* p) {
  return *reinterpret_cast<const bf16x8*>(p);
}
static __device__ __forceinline__ v8f wmma_bf16(v16bf a, v16bf b, v8f c) {
  // D = A(16x32 bf16) x B(32x16 bf16) + C(16x16 f32)
  return __builtin_amdgcn_wmma_f32_16x16x32_bf16(false, a, false, b,
                                                 (short)0, c, false, false);
}

// A-operand load: 16x32 bf16 slice, row-major source, base -> element [row0][col0].
// lane: row = lane%16; K chunks [8h,8h+8) and [16+8h,16+8h+8), h = lane/16.
static __device__ __forceinline__ v16bf load_a32(const bf16* base, int ld) {
  int lane = threadIdx.x & 31;
  int row = lane & 15, lh = lane >> 4;
  const bf16* p = base + row * ld + lh * 8;
  return cat8(ld8(p), ld8(p + 16));
}

// B-operand load: K=32 x N=16 slice from Bt (N-major: Bt[n][k] contiguous in k).
// lane: col = lane%16; K chunks [8h,8h+8) and [16+8h,16+8h+8), h = lane/16.
static __device__ __forceinline__ v16bf load_b32(const bf16* base, int ld) {
  int lane = threadIdx.x & 31;
  int col = lane & 15, lh = lane >> 4;
  const bf16* p = base + col * ld + lh * 8;
  return cat8(ld8(p), ld8(p + 16));
}

// --------------------------- conversion kernels ----------------------------

__global__ void k_cvt(const float* __restrict__ src, bf16* __restrict__ dst, int n) {
  int i = blockIdx.x * blockDim.x + threadIdx.x;
  if (i < n) dst[i] = (bf16)src[i];
}

// src rows x cols (row-major) -> dst cols x rows (transposed, row-major)
__global__ void k_cvt_t(const float* __restrict__ src, bf16* __restrict__ dst,
                        int rows, int cols) {
  int i = blockIdx.x * blockDim.x + threadIdx.x;
  if (i < rows * cols) {
    int c = i / rows, r = i - c * rows;
    dst[i] = (bf16)src[r * cols + c];
  }
}

// vw (N x HD) -> vwT (B,H,DH,S): column-of-V contiguous in key index.
__global__ void k_vT(const bf16* __restrict__ vw, bf16* __restrict__ vwT) {
  int i = blockIdx.x * blockDim.x + threadIdx.x;   // over N*HD = 4M
  int s  = i & (SS - 1);
  int t  = i >> 10;
  int dh = t & (DKK - 1);
  int bh = t >> 6;
  int h  = bh & (HH - 1);
  int b  = bh >> 4;
  vwT[i] = vw[((size_t)(b * SS + s)) * HD + h * DKK + dh];
}

// --------------------------- projection GEMM -------------------------------
// Y = A(M x 1024) @ W + bias.  BT is W transposed (1024 x 1024, bf16).
// Block = 128 threads (4 waves), 64x64 output tile, wave w does rows [16w,16w+16).
// If out1 != null: out0 = Y + r0[col], out1 = Y + r1[col] (q projection).
__global__ void k_proj(const bf16* __restrict__ A, const bf16* __restrict__ BT,
                       const float* __restrict__ bias,
                       const float* __restrict__ r0, const float* __restrict__ r1,
                       bf16* __restrict__ out0, bf16* __restrict__ out1) {
  int w = threadIdx.x >> 5;
  int m0 = blockIdx.x * 64 + w * 16;
  int n0 = blockIdx.y * 64;

  v8f acc[4];
#pragma unroll
  for (int g = 0; g < 4; ++g) acc[g] = zero_v8f();

  for (int k0 = 0; k0 < DD; k0 += 32) {
    if (k0 + 32 < DD) __builtin_prefetch(A + (size_t)m0 * DD + k0 + 32, 0, 0);
    v16bf a = load_a32(A + (size_t)m0 * DD + k0, DD);
#pragma unroll
    for (int g = 0; g < 4; ++g) {
      v16bf b = load_b32(BT + (size_t)(n0 + 16 * g) * DD + k0, DD);
      acc[g] = wmma_bf16(a, b, acc[g]);
    }
  }

  int lane = threadIdx.x & 31;
  int colb = lane & 15, lh = lane >> 4;
#pragma unroll
  for (int g = 0; g < 4; ++g) {
#pragma unroll
    for (int v = 0; v < 8; ++v) {
      int row = m0 + v + 8 * lh;
      int col = n0 + 16 * g + colb;
      float val = acc[g][v] + bias[col];
      if (out1) {
        out0[(size_t)row * HD + col] = (bf16)(val + r0[col]);
        out1[(size_t)row * HD + col] = (bf16)(val + r1[col]);
      } else {
        out0[(size_t)row * HD + col] = (bf16)val;
      }
    }
  }
}

// --------------------------- BD (rel) GEMM + shift -------------------------
// G = qv_bh (S x DK) @ qrel_h^T (DK x S); scatter each G[r,m] to its shifted
// position:  m >= S-1-r  ->  (j=r,   k=m-(S-1-r))
//            m <= S-2-r  ->  (j=r-1, k=m+r+1)   (dropped when r==0)
__global__ void k_bd_diag(bf16* __restrict__ BDs) {
  int i = blockIdx.x * blockDim.x + threadIdx.x;   // over B*H*S
  int j  = i & (SS - 1);
  int bh = i >> 10;
  if (bh < BB * HH && j + 1 < SS)
    BDs[((size_t)bh * SS + j) * SS + (j + 1)] = (bf16)0.0f;
}

__global__ void k_bd(const bf16* __restrict__ qv, const bf16* __restrict__ qrel,
                     bf16* __restrict__ BDs) {
  int bh = blockIdx.z;
  int b = bh >> 4, h = bh & (HH - 1);
  int r0 = blockIdx.x * 16, m0 = blockIdx.y * 16;

  const bf16* Abase = qv   + ((size_t)(b * SS + r0)) * HD + h * DKK;
  const bf16* Bbase = qrel + ((size_t)m0) * HD + h * DKK;

  v16bf a_lo = load_a32(Abase,      HD);
  v16bf a_hi = load_a32(Abase + 32, HD);
  v16bf b_lo = load_b32(Bbase,      HD);
  v16bf b_hi = load_b32(Bbase + 32, HD);

  v8f c = zero_v8f();
  c = wmma_bf16(a_lo, b_lo, c);
  c = wmma_bf16(a_hi, b_hi, c);

  int lane = threadIdx.x & 31;
  int colb = lane & 15, lh = lane >> 4;
  bf16* base = BDs + (size_t)bh * SS * SS;
#pragma unroll
  for (int v = 0; v < 8; ++v) {
    int r = r0 + v + 8 * lh;
    int m = m0 + colb;
    float val = c[v];
    if (m >= SS - 1 - r) {
      int j = r, k = m - (SS - 1 - r);
      base[(size_t)j * SS + k] = (bf16)val;
    } else if (r > 0) {
      int j = r - 1, k = m + r + 1;
      base[(size_t)j * SS + k] = (bf16)val;
    }
  }
}

// --------------------------- flash attention -------------------------------
// Block = 128 threads (4 waves). Wave w owns query rows [j0, j0+16),
// streams 32 keys per iteration: 4 WMMA scores, BDs gather, online softmax,
// P through LDS (re-layout C->A operand), 4 WMMA for P@V.
__global__ void k_attn(const bf16* __restrict__ qu, const bf16* __restrict__ kw,
                       const bf16* __restrict__ vwT, const bf16* __restrict__ BDs,
                       float* __restrict__ out) {
  __shared__ __align__(16) bf16 pbuf[4][16][32];

  int w = threadIdx.x >> 5;
  int lane = threadIdx.x & 31;
  int colb = lane & 15, lh = lane >> 4;
  int h = blockIdx.y, b = blockIdx.z;
  int j0 = blockIdx.x * 64 + w * 16;

  const bf16* Aq = qu + ((size_t)(b * SS + j0)) * HD + h * DKK;
  v16bf a_lo = load_a32(Aq,      HD);
  v16bf a_hi = load_a32(Aq + 32, HD);

  const bf16* Kb = kw  + (size_t)b * SS * HD + h * DKK;           // + k*HD
  const bf16* Vb = vwT + ((size_t)(b * HH + h)) * DKK * SS;       // + dh*S
  const bf16* Bd = BDs + ((size_t)(b * HH + h)) * SS * SS + (size_t)j0 * SS;

  float mrow[8], lrow[8];
  v8f o[4];
#pragma unroll
  for (int g = 0; g < 4; ++g) o[g] = zero_v8f();
#pragma unroll
  for (int v = 0; v < 8; ++v) { mrow[v] = -1e30f; lrow[v] = 0.0f; }

  const float scale = 0.125f;   // 1/sqrt(DK)

  for (int k0 = 0; k0 < SS; k0 += 32) {
    if (k0 + 32 < SS) __builtin_prefetch(Kb + (size_t)(k0 + 32) * HD, 0, 0);

    // --- scores: two 16-col subtiles, K-dim = DK = 64 (2 WMMA each) ---
    v8f s0 = zero_v8f(), s1 = zero_v8f();
    {
      v16bf bl0 = load_b32(Kb + (size_t)k0 * HD,      HD);
      v16bf bh0 = load_b32(Kb + (size_t)k0 * HD + 32, HD);
      s0 = wmma_bf16(a_lo, bl0, s0);
      s0 = wmma_bf16(a_hi, bh0, s0);
      v16bf bl1 = load_b32(Kb + (size_t)(k0 + 16) * HD,      HD);
      v16bf bh1 = load_b32(Kb + (size_t)(k0 + 16) * HD + 32, HD);
      s1 = wmma_bf16(a_lo, bl1, s1);
      s1 = wmma_bf16(a_hi, bh1, s1);
    }

    // --- add shifted-BD, scale ---
    float p0[8], p1[8];
#pragma unroll
    for (int v = 0; v < 8; ++v) {
      int j = v + 8 * lh;
      float bd0 = (float)Bd[(size_t)j * SS + k0 + colb];
      float bd1 = (float)Bd[(size_t)j * SS + k0 + 16 + colb];
      p0[v] = (s0[v] + bd0) * scale;
      p1[v] = (s1[v] + bd1) * scale;
    }

    // --- online softmax (rows striped across 16-lane halves) ---
#pragma unroll
    for (int v = 0; v < 8; ++v) {
      float mx = fmaxf(p0[v], p1[v]);
      mx = fmaxf(mx, __shfl_xor(mx, 1));
      mx = fmaxf(mx, __shfl_xor(mx, 2));
      mx = fmaxf(mx, __shfl_xor(mx, 4));
      mx = fmaxf(mx, __shfl_xor(mx, 8));
      float mnew  = fmaxf(mrow[v], mx);
      float alpha = __expf(mrow[v] - mnew);
      float e0 = __expf(p0[v] - mnew);
      float e1 = __expf(p1[v] - mnew);
      float rs = e0 + e1;
      rs += __shfl_xor(rs, 1);
      rs += __shfl_xor(rs, 2);
      rs += __shfl_xor(rs, 4);
      rs += __shfl_xor(rs, 8);
      lrow[v] = lrow[v] * alpha + rs;
      mrow[v] = mnew;
      p0[v] = e0;
      p1[v] = e1;
#pragma unroll
      for (int g = 0; g < 4; ++g) o[g][v] *= alpha;
    }

    // --- stage P (16x32, C-layout) into LDS; re-read in A-operand layout ---
#pragma unroll
    for (int v = 0; v < 8; ++v) {
      int j = v + 8 * lh;
      pbuf[w][j][colb]      = (bf16)p0[v];
      pbuf[w][j][16 + colb] = (bf16)p1[v];
    }
    __syncthreads();   // uniform trip count across all 4 waves

    v16bf pa;
    {
      const bf16* prow = &pbuf[w][colb][0];          // row = lane%16
      pa = cat8(ld8(prow + 8 * lh), ld8(prow + 16 + 8 * lh));
    }

    // --- out += P(16x32) @ V(32x64): 4 WMMA (one per 16-col dh group) ---
#pragma unroll
    for (int g = 0; g < 4; ++g) {
      v16bf bv = load_b32(Vb + (size_t)(16 * g) * SS + k0, SS);
      o[g] = wmma_bf16(pa, bv, o[g]);
    }
    __syncthreads();
  }

  // --- normalize and store fp32 output ---
#pragma unroll
  for (int v = 0; v < 8; ++v) {
    int j = j0 + v + 8 * lh;
    float inv = 1.0f / lrow[v];
#pragma unroll
    for (int g = 0; g < 4; ++g) {
      out[((size_t)(b * SS + j)) * HD + h * DKK + 16 * g + colb] = o[g][v] * inv;
    }
  }
}

// ---------------------------------------------------------------------------

extern "C" void kernel_launch(void* const* d_in, const int* in_sizes, int n_in,
                              void* d_out, int out_size, void* d_ws, size_t ws_size,
                              hipStream_t stream) {
  const float* x    = (const float*)d_in[0];
  const float* rel  = (const float*)d_in[1];
  const float* Wq   = (const float*)d_in[2];
  const float* bq   = (const float*)d_in[3];
  const float* Wk   = (const float*)d_in[4];
  const float* bk   = (const float*)d_in[5];
  const float* Wv   = (const float*)d_in[6];
  const float* bv   = (const float*)d_in[7];
  const float* Wp   = (const float*)d_in[8];
  const float* bp   = (const float*)d_in[9];
  const float* r_u  = (const float*)d_in[10];
  const float* r_v  = (const float*)d_in[11];
  float* out = (float*)d_out;

  char* ws = (char*)d_ws;
  const size_t MB = 1024ull * 1024ull;
  bf16* xbf   = (bf16*)(ws + 0);          //  8 MB : N x D
  bf16* relbf = (bf16*)(ws + 8  * MB);    //  2 MB : S x D
  bf16* WqT   = (bf16*)(ws + 10 * MB);    //  2 MB each, transposed (HD x D)
  bf16* WkT   = (bf16*)(ws + 12 * MB);
  bf16* WvT   = (bf16*)(ws + 14 * MB);
  bf16* WpT   = (bf16*)(ws + 16 * MB);
  bf16* qu    = (bf16*)(ws + 18 * MB);    //  8 MB : q + r_u
  bf16* qv    = (bf16*)(ws + 26 * MB);    //  8 MB : q + r_v
  bf16* kwv   = (bf16*)(ws + 34 * MB);    //  8 MB : keys
  bf16* vwv   = (bf16*)(ws + 42 * MB);    //  8 MB : values (N x HD)
  bf16* vwT   = (bf16*)(ws + 50 * MB);    //  8 MB : values (B,H,DH,S)
  bf16* qrel  = (bf16*)(ws + 58 * MB);    //  2 MB : rel projection (S x HD)
  bf16* BDs   = (bf16*)(ws + 60 * MB);    // 128 MB: shifted BD (B,H,S,S)

  // 1. fp32 -> bf16 conversions (weights transposed for B-operand reads)
  k_cvt  <<<(NN * DD) / 256, 256, 0, stream>>>(x,   xbf,   NN * DD);
  k_cvt  <<<(SS * DD) / 256, 256, 0, stream>>>(rel, relbf, SS * DD);
  k_cvt_t<<<(DD * HD) / 256, 256, 0, stream>>>(Wq, WqT, DD, HD);
  k_cvt_t<<<(DD * HD) / 256, 256, 0, stream>>>(Wk, WkT, DD, HD);
  k_cvt_t<<<(DD * HD) / 256, 256, 0, stream>>>(Wv, WvT, DD, HD);
  k_cvt_t<<<(DD * HD) / 256, 256, 0, stream>>>(Wp, WpT, DD, HD);

  // 2. projections (bf16 WMMA GEMM)
  dim3 gx(NN / 64, HD / 64), gr(SS / 64, HD / 64);
  k_proj<<<gx, 128, 0, stream>>>(xbf,   WqT, bq, r_u, r_v, qu, qv);
  k_proj<<<gx, 128, 0, stream>>>(xbf,   WkT, bk, nullptr, nullptr, kwv, nullptr);
  k_proj<<<gx, 128, 0, stream>>>(xbf,   WvT, bv, nullptr, nullptr, vwv, nullptr);
  k_proj<<<gr, 128, 0, stream>>>(relbf, WpT, bp, nullptr, nullptr, qrel, nullptr);

  // 3. transpose V for the PV B-operand
  k_vT<<<(NN * HD) / 256, 256, 0, stream>>>(vwv, vwT);

  // 4. BD GEMM + exact rel_shift scatter (diagonal zeros first)
  k_bd_diag<<<(BB * HH * SS) / 256, 256, 0, stream>>>(BDs);
  k_bd<<<dim3(SS / 16, SS / 16, BB * HH), 32, 0, stream>>>(qv, qrel, BDs);

  // 5. flash attention
  k_attn<<<dim3(SS / 64, HH, BB), 128, 0, stream>>>(qu, kwv, vwT, BDs, out);

  (void)in_sizes; (void)n_in; (void)out_size; (void)ws_size;
}
